// BatchAugmentation_73254962201022
// MI455X (gfx1250) — matrix operand order repaired and checked by reading, not compiled, not run on hardware
//
#include <hip/hip_runtime.h>
#include <hip/hip_bf16.h>

typedef __attribute__((ext_vector_type(16))) _Float16 v16h;
typedef __attribute__((ext_vector_type(8)))  float    v8f;
typedef __attribute__((ext_vector_type(4)))  float    f4;

#define D_GENES 4000
#define B_CELLS 4096
#define NCLS    32
#define ROWS_OUT (NCLS + 2 * B_CELLS)   /* 8224 */
#define TILE_G  64
#define PITCH   40                      /* halves per LDS gene row: 80B, 16B aligned */

__device__ __forceinline__ unsigned mix32(unsigned s) {
  s *= 0x9E3779B9u; s ^= s >> 16; s *= 0x85EBCA6Bu; s ^= s >> 13;
  return s;
}

// ---------------- metadata kernels ----------------

__global__ void init_ws_kernel(int* firstIdx, float* cnt) {
  int i = threadIdx.x;
  if (i < NCLS) { firstIdx[i] = 0x7FFFFFFF; cnt[i] = 0.0f; }
}

__global__ void scan_labels_kernel(const int* __restrict__ label,
                                   int* firstIdx, float* cnt) {
  int i = blockIdx.x * blockDim.x + threadIdx.x;
  if (i < B_CELLS) {
    int c = label[i];
    atomicMin(&firstIdx[c], i);
    atomicAdd(&cnt[c], 1.0f);
  }
}

__global__ void meta_kernel(const int* __restrict__ batch,
                            const int* __restrict__ label,
                            const float* __restrict__ cont,
                            const int* __restrict__ cat,
                            const int* __restrict__ firstIdx,
                            float* __restrict__ out) {
  int i = blockIdx.x * blockDim.x + threadIdx.x;
  if (i >= ROWS_OUT) return;
  size_t base = (size_t)ROWS_OUT * D_GENES;
  float* ob = out + base;                    // augmented_batch_index
  float* ol = ob + ROWS_OUT;                 // augmented_label
  float* oc = ol + ROWS_OUT;                 // augmented_cont (x4)
  float* oa = oc + (size_t)ROWS_OUT * 4;     // augmented_cat  (x2)
  int src; float lab;
  if (i < 2 * B_CELLS) { src = i & (B_CELLS - 1); lab = (float)label[src]; }
  else                 { int c = i - 2 * B_CELLS; src = firstIdx[c]; lab = (float)c; }
  ob[i] = (float)batch[src];
  ol[i] = lab;
  #pragma unroll
  for (int j = 0; j < 4; ++j) oc[(size_t)i * 4 + j] = cont[(size_t)src * 4 + j];
  #pragma unroll
  for (int j = 0; j < 2; ++j) oa[(size_t)i * 2 + j] = (float)cat[(size_t)src * 2 + j];
}

// ---------------- fused aug + one-hot-GEMM segment-sum kernel ----------------
// Workgroup = 128 threads (4 waves). Owns 64 gene columns. Streams all 4096
// rows exactly once: aug1/aug2 written from the loaded f32 tile; tile
// transposed into LDS as f16; per-class segment sum computed as a one-hot
// GEMM with v_wmma_f32_16x16x32_f16 (exact: x in {0..19}, one-hot in {0,1},
// f32 accumulation).

__global__ __launch_bounds__(128)
void aug_wmma_kernel(const float* __restrict__ x,
                     const int* __restrict__ label,
                     const float* __restrict__ cnt,
                     float* __restrict__ out) {
  __shared__ unsigned short s_label[B_CELLS];
  __shared__ __align__(16) _Float16 s_x[TILE_G * PITCH];

  const int tid  = threadIdx.x;
  const int lane = tid & 31;
  const int wave = tid >> 5;
  const int hi   = (lane >> 4) & 1;          // half-wave select
  const int m    = lane & 15;                // class-in-tile / gene-in-tile
  const int g0   = blockIdx.x * TILE_G;

  for (int i = tid; i < B_CELLS; i += 128)
    s_label[i] = (unsigned short)label[i];

  v8f acc0 = {};   // classes 0..15
  v8f acc1 = {};   // classes 16..31

  const int gq  = tid & 15;   // which float4 within the 64-gene row
  const int kr0 = tid >> 4;   // base row 0..7

  for (int kstep = 0; kstep < B_CELLS / 32; ++kstep) {
    const int k0 = kstep * 32;
    __syncthreads();          // previous B-fragment reads of s_x complete
    #pragma unroll
    for (int i = 0; i < 4; ++i) {
      const int kr = kr0 + i * 8;          // 0..31
      const int kg = k0 + kr;              // global cell row
      const int gc = g0 + gq * 4;          // global gene col (float4 base)
      const bool ok = (gc < D_GENES);
      f4 xv = {};
      if (ok) xv = *(const f4*)(x + (size_t)kg * D_GENES + gc);

      // per-row downsample rate in [0.7, 1.0); failure prob q <= 0.3
      float rate = 0.7f + 0.3f * ((float)mix32((unsigned)kg + 0x1234567u) *
                                  (1.0f / 4294967296.0f));
      int  tq = (int)((1.0f - rate) * 256.0f + 0.5f);       // 0..77 (<128)
      unsigned M = 0x01010101u * (unsigned)(127 + tq);      // SWAR threshold

      // one dropout hash per float4: byte j < 179 -> keep (P = 0.7)
      unsigned hd = mix32((unsigned)(kg * (D_GENES / 4) + (gc >> 2)) ^ 0x68BC21EBu);

      f4 o1, o2;
      #pragma unroll
      for (int j = 0; j < 4; ++j) {
        float keep = (((hd >> (j * 8)) & 0xFFu) < 179u) ? 1.0f : 0.0f;
        o1[j] = xv[j] * keep;

        // aug2: exact Bernoulli-sum binomial thinning, branchless SWAR.
        // 5 fixed groups of 4 byte-trials; MSB-per-failing-byte, popcount.
        int n = (int)xv[j];                                  // 0..19
        unsigned eseed = (unsigned)(kg * D_GENES + gc + j) * 0x9E3779B9u;
        int fails = 0;
        #pragma unroll
        for (int t = 0; t < 5; ++t) {
          unsigned h = mix32(eseed + (unsigned)t * 0x632BE5ABu);
          int v = n - t * 4;
          unsigned vv = (unsigned)(v < 0 ? 0 : (v > 4 ? 4 : v));
          unsigned msel = (vv == 0u) ? 0u : (0x80808080u >> ((4u - vv) * 8u));
          unsigned r = (M - (h & 0x7F7F7F7Fu)) & ~h & msel;
          fails += __popc(r);
        }
        o2[j] = (float)(n - fails);

        // transpose into LDS as f16 (exact for 0..19)
        s_x[(gq * 4 + j) * PITCH + kr] = (_Float16)xv[j];
      }
      if (ok) {
        *(f4*)(out + (size_t)(NCLS + kg) * D_GENES + gc)           = o1;
        *(f4*)(out + (size_t)(NCLS + B_CELLS + kg) * D_GENES + gc) = o2;
        if (kstep + 1 < B_CELLS / 32)
          __builtin_prefetch(x + (size_t)(kg + 32) * D_GENES + gc, 0, 1);
      }
    }
    __syncthreads();

    // A fragments: one-hot(label) 16x32 f16. lane M = lane&15;
    // half h -> K = (h<8 ? h : h+8) + 8*hi  (ISA 16-bit A layout)
    v16h a0, a1;
    #pragma unroll
    for (int h = 0; h < 16; ++h) {
      int kk = k0 + ((h < 8) ? h : h + 8) + hi * 8;
      int lv = (int)s_label[kk];
      a0[h] = (lv == m)        ? (_Float16)1.0f : (_Float16)0.0f;
      a1[h] = (lv == (m + 16)) ? (_Float16)1.0f : (_Float16)0.0f;
    }
    // B fragment: 32x16 f16, lane column N = lane&15; half h -> K = (h<8 ? h : h+8) + 8*hi
    const int colb = (wave << 4) + m;
    const int kb   = hi * 8;
    v16h bf;
    #pragma unroll
    for (int h = 0; h < 16; ++h) bf[h] = s_x[colb * PITCH + kb + ((h < 8) ? h : h + 8)];

    acc0 = __builtin_amdgcn_wmma_f32_16x16x32_f16(false, a0, false, bf,
                                                  (short)0, acc0, false, false);
    acc1 = __builtin_amdgcn_wmma_f32_16x16x32_f16(false, a1, false, bf,
                                                  (short)0, acc1, false, false);
  }

  // D layout: lane N = lane&15, VGPR r -> M = r + 8*hi. mean = clamp(round(sum/cnt), 0)
  const int gene = g0 + (wave << 4) + m;
  if (gene < D_GENES) {
    #pragma unroll
    for (int r = 0; r < 8; ++r) {
      int c0 = r + hi * 8;
      out[(size_t)c0 * D_GENES + gene] = fmaxf(rintf(acc0[r] / cnt[c0]), 0.0f);
      int c1 = c0 + 16;
      out[(size_t)c1 * D_GENES + gene] = fmaxf(rintf(acc1[r] / cnt[c1]), 0.0f);
    }
  }
}

// ---------------- launch ----------------

extern "C" void kernel_launch(void* const* d_in, const int* in_sizes, int n_in,
                              void* d_out, int out_size, void* d_ws, size_t ws_size,
                              hipStream_t stream) {
  (void)in_sizes; (void)n_in; (void)out_size; (void)ws_size;
  const float* x     = (const float*)d_in[0];
  const int*   batch = (const int*)d_in[1];
  const int*   label = (const int*)d_in[2];
  const float* cont  = (const float*)d_in[3];
  const int*   cat   = (const int*)d_in[4];
  float* out = (float*)d_out;

  int*   firstIdx = (int*)d_ws;
  float* cnt      = (float*)((char*)d_ws + NCLS * sizeof(int));

  init_ws_kernel<<<1, 32, 0, stream>>>(firstIdx, cnt);
  scan_labels_kernel<<<(B_CELLS + 255) / 256, 256, 0, stream>>>(label, firstIdx, cnt);
  meta_kernel<<<(ROWS_OUT + 255) / 256, 256, 0, stream>>>(batch, label, cont, cat,
                                                          firstIdx, out);
  aug_wmma_kernel<<<(D_GENES + TILE_G - 1) / TILE_G, 128, 0, stream>>>(x, label, cnt, out);
}
